// ShakespeareLeafNet_52785148067936
// MI455X (gfx1250) — matrix operand order repaired and checked by reading, not compiled, not run on hardware
//
#include <hip/hip_runtime.h>
#include <math.h>

// ---------------------------------------------------------------------------
// ShakespeareLeafNet on MI455X (gfx1250): 2-layer LSTM + decoder.
// f16 WMMA (v_wmma_f32_16x16x32_f16) for all GEMMs, f32 accumulate,
// fused LSTM pointwise. Each wave: 32 batch rows x 16 hidden units x 4 gates.
// Layer0 input (K=8) is zero-padded to K=32 at embed time so both layers use
// the same branch-free strided fragment loads.
// ---------------------------------------------------------------------------

#define T_LEN 80
#define BATCH 5120
#define HID   256
#define EMB   8
#define EMBP  32          // padded K for layer0 input GEMM
#define VOCAB 100
#define G4H   (4 * HID)   // 1024
#define MT    2           // 16-row tiles per wave (M = 32 per wave)

typedef _Float16 f16;
typedef __attribute__((ext_vector_type(16))) _Float16 v16h;
typedef __attribute__((ext_vector_type(8)))  _Float16 v8h;
typedef __attribute__((ext_vector_type(8)))  float    v8f;

union AFrag { v16h v; struct { v8h lo; v8h hi; } c; };

// ---- WMMA fragment loads (layouts per CDNA5 ISA 7.12.2, wave32) ------------

// A matrix 16x32 f16, row-major source, leading dim ldk (halves).
// lane L: row = mbase + (L&15); elems 0..7  = K[kstart + (L>>4)*8 ..+7],
//         elems 8..15 = K[kstart + 16 + (L>>4)*8 ..+7]
__device__ inline v16h load_a_frag(const f16* __restrict__ base, int ldk,
                                   int kstart, int mbase, int lane) {
    const int m  = mbase + (lane & 15);
    const int kh = lane >> 4;
    const f16* p = base + (size_t)m * ldk + kstart + kh * 8;
    AFrag a;
    a.c.lo = *(const v8h*)p;
    a.c.hi = *(const v8h*)(p + 16);
    return a.v;
}

// B matrix 32x16 f16 where B[k][n] = W[n][k]; W row-major, leading dim ldk.
// lane L: col n = nbase + (L&15); elems 0..7  = K[kstart + (L>>4)*8 ..+7],
//         elems 8..15 = K[kstart + 16 + (L>>4)*8 ..+7]
__device__ inline v16h load_b_frag(const f16* __restrict__ w, int ldk,
                                   int kstart, int nbase, int lane) {
    const int n  = nbase + (lane & 15);
    const int kh = lane >> 4;
    const f16* p = w + (size_t)n * ldk + kstart + kh * 8;
    AFrag b;
    b.c.lo = *(const v8h*)p;
    b.c.hi = *(const v8h*)(p + 16);
    return b.v;
}

__device__ inline v8f wmma_f16(v16h a, v16h b, v8f c) {
    return __builtin_amdgcn_wmma_f32_16x16x32_f16(false, a, false, b,
                                                  (short)0, c, false, false);
}

__device__ inline float sigmoidf_(float x) { return 1.f / (1.f + __expf(-x)); }

// ---------------------------------------------------------------------------
// Fused LSTM step. KX = EMBP (layer0, 1 K-chunk) or HID (layer1, 8 K-chunks).
// h_prev == nullptr / c_prev == nullptr -> t = 0 (zero state).
// ---------------------------------------------------------------------------
template<int KX>
__global__ __launch_bounds__(256)
void lstm_step_kernel(const f16* __restrict__ x_t,     // [B x KX]
                      const f16* __restrict__ wx,      // [4H x KX] f16
                      const f16* __restrict__ h_prev,  // [B x H] f16 or null
                      const f16* __restrict__ wh,      // [4H x H] f16
                      const float* __restrict__ b_ih,
                      const float* __restrict__ b_hh,
                      const float* __restrict__ c_prev, // [B x H] or null
                      float* __restrict__ c_out,        // [B x H]
                      f16*   __restrict__ h_out,        // [B x H]
                      float* __restrict__ h_out32)      // [B x H] or null
{
    const int lane   = threadIdx.x & 31;
    const int wid    = (blockIdx.x * blockDim.x + threadIdx.x) >> 5;
    const int ntiles = HID / 16;                       // 16
    const int mbase  = (wid / ntiles) * (16 * MT);     // 160 m-blocks
    const int nbase  = (wid % ntiles) * 16;

    v8f zero = {};
    v8f acc[4][MT];
    #pragma unroll
    for (int g = 0; g < 4; ++g)
        #pragma unroll
        for (int mt = 0; mt < MT; ++mt) acc[g][mt] = zero;

    // ---- input contribution (branch-free; layer0 is zero-padded to K=32) ----
    #pragma unroll
    for (int kc = 0; kc < KX; kc += 32) {
        v16h a[MT];
        #pragma unroll
        for (int mt = 0; mt < MT; ++mt)
            a[mt] = load_a_frag(x_t, KX, kc, mbase + 16 * mt, lane);
        #pragma unroll
        for (int g = 0; g < 4; ++g) {
            v16h b = load_b_frag(wx, KX, kc, g * HID + nbase, lane);
            #pragma unroll
            for (int mt = 0; mt < MT; ++mt)
                acc[g][mt] = wmma_f16(a[mt], b, acc[g][mt]);
        }
    }

    // ---- recurrent contribution ----
    if (h_prev != nullptr) {
        #pragma unroll
        for (int kc = 0; kc < HID; kc += 32) {
            v16h a[MT];
            #pragma unroll
            for (int mt = 0; mt < MT; ++mt)
                a[mt] = load_a_frag(h_prev, HID, kc, mbase + 16 * mt, lane);
            #pragma unroll
            for (int g = 0; g < 4; ++g) {
                v16h b = load_b_frag(wh, HID, kc, g * HID + nbase, lane);
                #pragma unroll
                for (int mt = 0; mt < MT; ++mt)
                    acc[g][mt] = wmma_f16(a[mt], b, acc[g][mt]);
            }
        }
    }

    // ---- fused LSTM cell update ----
    // C/D layout: elem r of lane L -> row tilebase + r + 8*(L>>4), col nbase+(L&15)
    const int n  = lane & 15;
    const int kh = lane >> 4;
    const int u  = nbase + n;
    const float bi = b_ih[0 * HID + u] + b_hh[0 * HID + u];
    const float bf = b_ih[1 * HID + u] + b_hh[1 * HID + u];
    const float bg = b_ih[2 * HID + u] + b_hh[2 * HID + u];
    const float bo = b_ih[3 * HID + u] + b_hh[3 * HID + u];

    #pragma unroll
    for (int mt = 0; mt < MT; ++mt) {
        #pragma unroll
        for (int r = 0; r < 8; ++r) {
            const int brow = mbase + 16 * mt + r + 8 * kh;
            const size_t idx = (size_t)brow * HID + u;
            const float cp = (c_prev != nullptr) ? c_prev[idx] : 0.f;
            const float gi = acc[0][mt][r] + bi;
            const float gf = acc[1][mt][r] + bf;
            const float gg = acc[2][mt][r] + bg;
            const float go = acc[3][mt][r] + bo;
            const float cn = sigmoidf_(gf) * cp + sigmoidf_(gi) * tanhf(gg);
            const float hn = sigmoidf_(go) * tanhf(cn);
            c_out[idx] = cn;
            h_out[idx] = (f16)hn;
            if (h_out32 != nullptr) h_out32[idx] = hn;
        }
    }
}

// ---------------------------------------------------------------------------
// Embedding lookup -> f16 rows padded to 32 halves (cols 8..31 = 0)
// ---------------------------------------------------------------------------
__global__ void embed_kernel(const int* __restrict__ x,
                             const float* __restrict__ emb,
                             f16* __restrict__ enc, int n)
{
    int i = blockIdx.x * blockDim.x + threadIdx.x;
    if (i >= n) return;
    const int idx = x[i];
    const float4* e = (const float4*)(emb + (size_t)idx * EMB);
    float4 e0 = e[0], e1 = e[1];
    v8h o;
    o[0] = (f16)e0.x; o[1] = (f16)e0.y; o[2] = (f16)e0.z; o[3] = (f16)e0.w;
    o[4] = (f16)e1.x; o[5] = (f16)e1.y; o[6] = (f16)e1.z; o[7] = (f16)e1.w;
    v8h z = {};
    v8h* d = (v8h*)(enc + (size_t)i * EMBP);
    d[0] = o; d[1] = z; d[2] = z; d[3] = z;
}

__global__ void f32_to_f16_kernel(const float* __restrict__ s,
                                  f16* __restrict__ d, int n)
{
    int i = blockIdx.x * blockDim.x + threadIdx.x;
    if (i < n) d[i] = (f16)s[i];
}

// Pad-convert w_ih0 [4H x 8] f32 -> [4H x 32] f16 (cols 8..31 = 0)
__global__ void pad_wih0_kernel(const float* __restrict__ s,
                                f16* __restrict__ d, int n /* = 4H*EMBP */)
{
    int i = blockIdx.x * blockDim.x + threadIdx.x;
    if (i >= n) return;
    const int row = i / EMBP;
    const int col = i % EMBP;
    d[i] = (col < EMB) ? (f16)s[(size_t)row * EMB + col] : (f16)0.f;
}

// ---------------------------------------------------------------------------
// Decoder: out[64,100] = flat(h1_last)[64,20480] @ dec_w^T + dec_b
// flat is the contiguous [5120,256] f32 buffer viewed as [64,20480].
// ---------------------------------------------------------------------------
__global__ __launch_bounds__(256)
void decoder_kernel(const float* __restrict__ flat,
                    const float* __restrict__ dec_w,
                    const float* __restrict__ dec_b,
                    float* __restrict__ out)
{
    const int wid  = (blockIdx.x * blockDim.x + threadIdx.x) >> 5;
    const int lane = threadIdx.x & 31;
    const int r = wid / VOCAB;          // 0..63
    const int v = wid % VOCAB;          // 0..99
    const int K = HID * T_LEN;          // 20480
    const float* a = flat  + (size_t)r * K;
    const float* w = dec_w + (size_t)v * K;
    float s = 0.f;
    for (int k = lane; k < K; k += 32) s += a[k] * w[k];
    #pragma unroll
    for (int off = 16; off > 0; off >>= 1) s += __shfl_down(s, off, 32);
    if (lane == 0) out[(size_t)r * VOCAB + v] = s + dec_b[v];
}

// ---------------------------------------------------------------------------
extern "C" void kernel_launch(void* const* d_in, const int* in_sizes, int n_in,
                              void* d_out, int out_size, void* d_ws, size_t ws_size,
                              hipStream_t stream)
{
    (void)in_sizes; (void)n_in; (void)out_size; (void)ws_size;

    const int*   x     = (const int*)  d_in[0];
    const float* emb   = (const float*)d_in[1];
    const float* w_ih0 = (const float*)d_in[2];
    const float* w_hh0 = (const float*)d_in[3];
    const float* b_ih0 = (const float*)d_in[4];
    const float* b_hh0 = (const float*)d_in[5];
    const float* w_ih1 = (const float*)d_in[6];
    const float* w_hh1 = (const float*)d_in[7];
    const float* b_ih1 = (const float*)d_in[8];
    const float* b_hh1 = (const float*)d_in[9];
    const float* dec_w = (const float*)d_in[10];
    const float* dec_b = (const float*)d_in[11];
    float* out = (float*)d_out;

    // ---- workspace carve-out (256B aligned) ----
    char* base = (char*)d_ws;
    size_t off = 0;
    auto carve = [&](size_t bytes) -> void* {
        void* p = base + off;
        off += (bytes + 255) & ~(size_t)255;
        return p;
    };
    f16*   enc    = (f16*)  carve((size_t)T_LEN * BATCH * EMBP * sizeof(f16)); // 25 MB
    f16*   wih0h  = (f16*)  carve((size_t)G4H * EMBP * sizeof(f16));
    f16*   whh0h  = (f16*)  carve((size_t)G4H * HID * sizeof(f16));
    f16*   wih1h  = (f16*)  carve((size_t)G4H * HID * sizeof(f16));
    f16*   whh1h  = (f16*)  carve((size_t)G4H * HID * sizeof(f16));
    f16*   ys0    = (f16*)  carve((size_t)T_LEN * BATCH * HID * sizeof(f16));  // 200 MB
    float* c0     = (float*)carve((size_t)BATCH * HID * sizeof(float));
    float* c1     = (float*)carve((size_t)BATCH * HID * sizeof(float));
    f16*   h1a    = (f16*)  carve((size_t)BATCH * HID * sizeof(f16));
    f16*   h1b    = (f16*)  carve((size_t)BATCH * HID * sizeof(f16));
    float* h1f32  = (float*)carve((size_t)BATCH * HID * sizeof(float));

    // ---- weight conversion f32 -> f16 ----
    {
        int n;
        n = G4H * EMBP; pad_wih0_kernel<<<(n + 255) / 256, 256, 0, stream>>>(w_ih0, wih0h, n);
        n = G4H * HID;  f32_to_f16_kernel<<<(n + 255) / 256, 256, 0, stream>>>(w_hh0, whh0h, n);
        n = G4H * HID;  f32_to_f16_kernel<<<(n + 255) / 256, 256, 0, stream>>>(w_ih1, wih1h, n);
        n = G4H * HID;  f32_to_f16_kernel<<<(n + 255) / 256, 256, 0, stream>>>(w_hh1, whh1h, n);
    }

    // ---- embedding (padded to K=32) ----
    {
        const int n = T_LEN * BATCH;
        embed_kernel<<<(n + 255) / 256, 256, 0, stream>>>(x, emb, enc, n);
    }

    // waves per step = (BATCH/(16*MT)) * (HID/16) = 160*16 = 2560 -> 320 blocks
    const int step_blocks = (BATCH / (16 * MT)) * (HID / 16) / 8;

    // ---- layer 0 scan (padded K=32 input, h stored into ys0) ----
    for (int t = 0; t < T_LEN; ++t) {
        const f16* xt = enc + (size_t)t * BATCH * EMBP;
        const f16* hp = (t == 0) ? nullptr : ys0 + (size_t)(t - 1) * BATCH * HID;
        const float* cp = (t == 0) ? nullptr : c0;
        f16* ho = ys0 + (size_t)t * BATCH * HID;
        lstm_step_kernel<EMBP><<<step_blocks, 256, 0, stream>>>(
            xt, wih0h, hp, whh0h, b_ih0, b_hh0, cp, c0, ho, nullptr);
    }

    // ---- layer 1 scan (K=256 input from ys0, ping-pong h) ----
    for (int t = 0; t < T_LEN; ++t) {
        const f16* xt = ys0 + (size_t)t * BATCH * HID;
        const f16* hp = (t == 0) ? nullptr : ((t & 1) ? h1a : h1b);
        const float* cp = (t == 0) ? nullptr : c1;
        f16* ho = (t & 1) ? h1b : h1a;
        float* h32 = (t == T_LEN - 1) ? h1f32 : nullptr;
        lstm_step_kernel<HID><<<step_blocks, 256, 0, stream>>>(
            xt, wih1h, hp, whh1h, b_ih1, b_hh1, cp, c1, ho, h32);
    }

    // ---- decoder: 64*100 outputs, one wave each -> 800 blocks of 8 waves ----
    decoder_kernel<<<(64 * VOCAB) / 8, 256, 0, stream>>>(h1f32, dec_w, dec_b, out);
}
